// GraphAttentionLayer_5677946765717
// MI455X (gfx1250) — matrix operand order repaired and checked by reading, not compiled, not run on hardware
//
#include <hip/hip_runtime.h>
#include <hip/hip_bf16.h>

typedef __attribute__((ext_vector_type(16))) __bf16 v16bf;
typedef __attribute__((ext_vector_type(8)))  float  v8f;

constexpr int   GN     = 8192;   // nodes
constexpr int   GIN    = 256;    // input features
constexpr int   GOUT   = 64;     // output features
constexpr float GALPHA = 0.2f;   // LeakyReLU slope

__device__ __forceinline__ float leakyf(float x) { return x > 0.f ? x : GALPHA * x; }

// ---------------------------------------------------------------------------
// Kernel A: h = feature @ W  (bf16 WMMA, f32 accum)
// Writes: hT (bf16, [GOUT][GN] transposed for B-operand loads),
//         s = h@a1, t = h@a2 (f32).
// ---------------------------------------------------------------------------
__global__ __launch_bounds__(128) void gat_h_kernel(
    const float* __restrict__ feature, const float* __restrict__ W,
    const float* __restrict__ a1, const float* __restrict__ a2,
    __bf16* __restrict__ hT, float* __restrict__ s_out, float* __restrict__ t_out)
{
  __shared__ __bf16 lds_WT[GOUT][GIN + 16];   // W transposed, bf16 (padded)
  __shared__ float  lds_hp[4][16][GOUT];      // per-wave partial h
  __shared__ float  lds_h[16][GOUT];          // reduced h (f32)

  const int tid   = threadIdx.x;
  const int wave  = tid >> 5;
  const int lane  = tid & 31;
  const int r     = lane & 15;   // row within 16-row block (A/C layout)
  const int khalf = lane >> 4;   // which K half this lane holds
  const int row0  = blockIdx.x * 16;
  const int rowG  = row0 + r;

  for (int i = tid; i < GIN * GOUT; i += 128) {
    int k = i >> 6, c = i & 63;
    lds_WT[c][k] = (__bf16)W[i];
  }
  __syncthreads();

  v8f acc[4] = {v8f{}, v8f{}, v8f{}, v8f{}};
  const float* frow = feature + (size_t)rowG * GIN;

  for (int kt = wave; kt < GIN / 32; kt += 4) {
    const int k0 = kt * 32;
    v16bf a;
    #pragma unroll
    for (int g = 0; g < 2; ++g) {
      const int kk = khalf * 16 + g * 8;
      float tf[8];
      *(float4*)&tf[0] = *(const float4*)(frow + k0 + kk);
      *(float4*)&tf[4] = *(const float4*)(frow + k0 + kk + 4);
      #pragma unroll
      for (int q = 0; q < 8; ++q) a[g * 8 + q] = (__bf16)tf[q];
    }
    const int kb = k0 + khalf * 16;
    #pragma unroll
    for (int slc = 0; slc < 4; ++slc) {
      v16bf b = *(const v16bf*)&lds_WT[slc * 16 + r][kb];
      acc[slc] = __builtin_amdgcn_wmma_f32_16x16x32_bf16(
          false, a, false, b, (short)0, acc[slc], false, false);
    }
  }

  #pragma unroll
  for (int slc = 0; slc < 4; ++slc)
    #pragma unroll
    for (int v = 0; v < 8; ++v)
      lds_hp[wave][khalf * 8 + v][slc * 16 + r] = acc[slc][v];
  __syncthreads();

  {
    const int idx0 = tid * 8;
    const int orow = idx0 >> 6;
    const int ocol = idx0 & 63;
    #pragma unroll
    for (int q = 0; q < 8; ++q) {
      float hv = lds_hp[0][orow][ocol + q] + lds_hp[1][orow][ocol + q]
               + lds_hp[2][orow][ocol + q] + lds_hp[3][orow][ocol + q];
      lds_h[orow][ocol + q] = hv;
      hT[(size_t)(ocol + q) * GN + row0 + orow] = (__bf16)hv;
    }
  }
  __syncthreads();

  if (tid < 32) {
    const int rr = tid & 15;
    const float* av = (tid < 16) ? a1 : a2;
    float d = 0.f;
    #pragma unroll
    for (int c = 0; c < GOUT; ++c) d = fmaf(lds_h[rr][c], av[c], d);
    ((tid < 16) ? s_out : t_out)[row0 + rr] = d;
  }
}

// ---------------------------------------------------------------------------
// Kernel P: T = max(t); E[j] = { exp(t_j - T), exp(0.2*(t_j - T)) }.
// Hoists ALL softmax transcendentals out of the N x N loop (LeakyReLU is
// piecewise linear, so exp factorizes per branch). Both factors are in (0,1].
// ---------------------------------------------------------------------------
__global__ __launch_bounds__(256) void gat_prep_kernel(
    const float* __restrict__ t, float* __restrict__ Tmax, float2* __restrict__ E)
{
  __shared__ float red[256];
  float m = -3.4e38f;
  for (int i = threadIdx.x; i < GN; i += 256) m = fmaxf(m, t[i]);
  red[threadIdx.x] = m;
  __syncthreads();
  for (int s = 128; s > 0; s >>= 1) {
    if (threadIdx.x < s) red[threadIdx.x] = fmaxf(red[threadIdx.x], red[threadIdx.x + s]);
    __syncthreads();
  }
  const float T = red[0];
  if (threadIdx.x == 0) Tmax[0] = T;
  for (int i = threadIdx.x; i < GN; i += 256) {
    const float x = t[i] - T;
    E[i] = make_float2(__expf(x), __expf(GALPHA * x));
  }
}

// ---------------------------------------------------------------------------
// Kernel B: fused masked-softmax attention + aggregation + ELU.
// One pass over adj (256 MB total -> 23.3 TB/s HBM roofline ~11us).
// Inner loop per adj element is exp-free:
//   p = adj * ( (E1_j > exp(-(s_i+T))) ? c1*E1_j : c2*E2_j )
// with per-row constants c1 = exp(s_i+T-m_i), c2 = exp(0.2(s_i+T)-m_i),
// m_i = lrelu(s_i+T) an upper bound on the true masked row max.
// ---------------------------------------------------------------------------
__global__ __launch_bounds__(128) void gat_attn_kernel(
    const float* __restrict__ adj, const float* __restrict__ s,
    const float* __restrict__ Tmax, const float2* __restrict__ E,
    const __bf16* __restrict__ hT, float* __restrict__ out)
{
  __shared__ float2 lds_e[GN];                // 64 KB: {E1, E2} per node
  __shared__ float  lds_acc[4][16][GOUT];     // per-wave partial outputs
  __shared__ float  lds_rowsum[16];           // softmax denominators

  const int tid   = threadIdx.x;
  const int wave  = tid >> 5;
  const int lane  = tid & 31;
  const int r     = lane & 15;
  const int khalf = lane >> 4;
  const int row0  = blockIdx.x * 16;
  const int rowG  = row0 + r;

  for (int i = tid; i < GN / 2; i += 128)     // 64 KB as float4 chunks
    ((float4*)lds_e)[i] = ((const float4*)E)[i];
  if (tid < 16) lds_rowsum[tid] = 0.f;
  __syncthreads();

  const float s_i  = s[rowG];
  const float sT   = s_i + Tmax[0];
  const float mrow = leakyf(sT);
  const float c1   = __expf(sT - mrow);           // <= 1
  const float c2   = __expf(GALPHA * sT - mrow);  // <= 1
  const float thr1 = __expf(-sT);                 // E1 > thr1  <=>  s_i+t_j > 0

  v8f acc[4] = {v8f{}, v8f{}, v8f{}, v8f{}};
  float lsum = 0.f;
  const float* adjrow = adj + (size_t)rowG * GN;

  for (int jt = wave; jt < GN / 32; jt += 4) {
    const int j0 = jt * 32;
    __builtin_prefetch(adjrow + j0 + 512, 0, 0);  // ~4 iters ahead of this wave

    // Build A operand (attention probs), contiguous K order (matches B):
    // lanes 0-15 hold K in {0..15}; lanes 16-31 hold {16..31}.
    v16bf a;
    #pragma unroll
    for (int g = 0; g < 2; ++g) {
      const int kk = khalf * 16 + g * 8;
      float adv[8];
      *(float4*)&adv[0] = *(const float4*)(adjrow + j0 + kk);
      *(float4*)&adv[4] = *(const float4*)(adjrow + j0 + kk + 4);
      #pragma unroll
      for (int q = 0; q < 8; ++q) {
        const float2 e12 = lds_e[j0 + kk + q];    // broadcast across lanes
        const float m1 = c1 * e12.x;
        const float m2 = c2 * e12.y;
        float p = (e12.x > thr1) ? m1 : m2;       // LeakyReLU branch select
        p *= adv[q];                              // adjacency mask (0/1)
        lsum += p;
        a[g * 8 + q] = (__bf16)p;
      }
    }

    // B operand: hT rows (feature columns), contiguous K in memory.
    const int jb = j0 + khalf * 16;
    #pragma unroll
    for (int slc = 0; slc < 4; ++slc) {
      v16bf b = *(const v16bf*)(hT + (size_t)(slc * 16 + r) * GN + jb);
      acc[slc] = __builtin_amdgcn_wmma_f32_16x16x32_bf16(
          false, a, false, b, (short)0, acc[slc], false, false);
    }
  }

  atomicAdd(&lds_rowsum[r], lsum);

  #pragma unroll
  for (int slc = 0; slc < 4; ++slc)
    #pragma unroll
    for (int v = 0; v < 8; ++v)
      lds_acc[wave][khalf * 8 + v][slc * 16 + r] = acc[slc][v];
  __syncthreads();

  {
    const int idx0 = tid * 8;
    const int orow = idx0 >> 6;
    const int ocol = idx0 & 63;
    const float inv = 1.0f / lds_rowsum[orow];
    #pragma unroll
    for (int q = 0; q < 8; ++q) {
      float v = lds_acc[0][orow][ocol + q] + lds_acc[1][orow][ocol + q]
              + lds_acc[2][orow][ocol + q] + lds_acc[3][orow][ocol + q];
      v *= inv;
      v = v > 0.f ? v : (__expf(v) - 1.0f);       // ELU (alpha=1)
      out[(size_t)(row0 + orow) * GOUT + ocol + q] = v;
    }
  }
}

// ---------------------------------------------------------------------------
extern "C" void kernel_launch(void* const* d_in, const int* in_sizes, int n_in,
                              void* d_out, int out_size, void* d_ws, size_t ws_size,
                              hipStream_t stream) {
  const float* feature = (const float*)d_in[0];   // [8192,256]
  const float* adj     = (const float*)d_in[1];   // [8192,8192]
  const float* W       = (const float*)d_in[2];   // [256,64]
  const float* a1      = (const float*)d_in[3];   // [64]
  const float* a2      = (const float*)d_in[4];   // [64]
  float*       out     = (float*)d_out;           // [8192,64]

  char* ws = (char*)d_ws;
  const size_t HT_BYTES = (size_t)GOUT * GN * sizeof(__bf16);       // 1 MB
  __bf16* hT   = (__bf16*)ws;
  float*  sbuf = (float*)(ws + HT_BYTES);                           // 32 KB
  float*  tbuf = (float*)(ws + HT_BYTES + (size_t)GN * 4);          // 32 KB
  float*  tmax = (float*)(ws + HT_BYTES + (size_t)2 * GN * 4);      // 64 B pad
  float2* Etab = (float2*)(ws + HT_BYTES + (size_t)2 * GN * 4 + 64);// 64 KB

  gat_h_kernel<<<GN / 16, 128, 0, stream>>>(feature, W, a1, a2, hT, sbuf, tbuf);
  gat_prep_kernel<<<1, 256, 0, stream>>>(tbuf, tmax, Etab);
  gat_attn_kernel<<<GN / 16, 128, 0, stream>>>(adj, sbuf, tmax, Etab, hT, out);
}
